// SlidingWindowAttention_43980465111627
// MI455X (gfx1250) — matrix operand order repaired and checked by reading, not compiled, not run on hardware
//
#include <hip/hip_runtime.h>

// ---------------- problem constants ----------------
#define B_   2
#define T_   2048
#define C_   1024
#define NH_  16
#define HS_  64
#define WIN_ 256
#define C3_  (3 * C_)
#define M_   (B_ * T_)      // 4096 rows

typedef __attribute__((ext_vector_type(16))) _Float16 v16h;
typedef __attribute__((ext_vector_type(8)))  _Float16 v8h;
typedef __attribute__((ext_vector_type(8)))  float    v8f;
typedef __attribute__((ext_vector_type(4)))  unsigned int u32x4;
typedef __attribute__((ext_vector_type(8)))  int          i32x8;
typedef __attribute__((ext_vector_type(4)))  int          i32x4;

__device__ __forceinline__ v16h cat8(v8h a, v8h b) {
  return __builtin_shufflevector(a, b, 0,1,2,3,4,5,6,7,8,9,10,11,12,13,14,15);
}
__device__ __forceinline__ v8f wmma16(v16h a, v16h b, v8f c) {
  // (neg_a, A, neg_b, B, c_mod, C, reuse_a, reuse_b)
  return __builtin_amdgcn_wmma_f32_16x16x32_f16(false, a, false, b, (short)0, c,
                                                false, false);
}

// ---- Tensor Data Mover: 2D f16 tile (tile_dim0 x tile_dim1) global -> LDS,
// with LDS row padding: pad_interval=16 DWORDs (64B row), pad_amount=4 DWORDs
// (16B) => LDS pitch = 40 f16, matching the fragment-read layout below.
__device__ __forceinline__ void tdm_load_tile_f16(const _Float16* gptr,
                                                  unsigned lds_off,
                                                  unsigned tile_d0, unsigned tile_d1,
                                                  unsigned row_stride_elems) {
  const unsigned long long ga = (unsigned long long)(uintptr_t)gptr;
  u32x4 g0;
  g0[0] = 1u;                                            // count=1 (valid D#)
  g0[1] = lds_off;                                       // lds_addr [63:32]
  g0[2] = (unsigned)(ga & 0xffffffffu);                  // global_addr [95:64]
  g0[3] = (unsigned)((ga >> 32) & 0x01ffffffu)           // global_addr [120:96]
          | (2u << 30);                                  // type=2 ("image") [127:126]
  i32x8 g1;
  g1[0] = (int)((1u << 16)      // data_size = 1 -> 2 bytes
                | (1u << 20)    // pad_enable
                | (3u << 22)    // pad_interval: 16 DWORDs (64B) before pad
                | (3u << 25));  // pad_amount: 4 DWORDs (16B)
  g1[1] = (int)(tile_d0 << 16);            // tensor_dim0[15:0] in [79:64]->word1 hi
  g1[2] = (int)(tile_d1 << 16);            // tensor_dim0 hi=0 | tensor_dim1[15:0]
  g1[3] = (int)(tile_d0 << 16);            // tensor_dim1 hi=0 | tile_dim0
  g1[4] = (int)tile_d1;                    // tile_dim1 | tile_dim2=0
  g1[5] = (int)row_stride_elems;           // tensor_dim0_stride[31:0]
  g1[6] = 0;                               // stride hi | tensor_dim1_stride lo
  g1[7] = 0;
  const i32x4 z4 = {0, 0, 0, 0};
#if __clang_major__ >= 23
  const i32x8 z8 = {0, 0, 0, 0, 0, 0, 0, 0};
  __builtin_amdgcn_tensor_load_to_lds(g0, g1, z4, z4, z8, 0);
#else
  __builtin_amdgcn_tensor_load_to_lds(g0, g1, z4, z4, 0);
#endif
}

// ---------------- f32 -> f16 conversion ----------------
__global__ void cvt_kernel(const float* __restrict__ src,
                           _Float16* __restrict__ dst, int n) {
  for (int i = blockIdx.x * blockDim.x + threadIdx.x; i < n;
       i += gridDim.x * blockDim.x)
    dst[i] = (_Float16)src[i];
}

// ---------------- tiled WMMA GEMM: Out = A(MxK) * B(KxN) ----------------
// f16 in, f32 accumulate, f16 or f32 out.  Block = 256 threads = 8 waves.
// Tile 128x128, K-chunk 32. Wave grid 4(M) x 2(N); each wave 32x64 = 2x4 WMMA
// tiles. A tile is DMA'd by the Tensor Data Mover (double-buffered, issued one
// chunk ahead by wave 0, tracked with TENSORcnt); B tile is staged transposed.
#define BM 128
#define BN 128
#define BK 32
#define LP (BK + 8)   // LDS pitch in f16 (80B rows -> conflict-reduced, 16B aligned)

__global__ __launch_bounds__(256) void gemm_kernel(
    const _Float16* __restrict__ A, const _Float16* __restrict__ Bm,
    void* __restrict__ Out, int M, int N, int K, int out_f16) {
  __shared__ __align__(16) _Float16 As[2][BM][LP];  // [buf][m][k], TDM-filled
  __shared__ __align__(16) _Float16 Bs[BN][LP];     // transposed: [n][k]

  const int tid  = threadIdx.x;
  const int lane = tid & 31;
  const int wave = tid >> 5;
  const int wm   = wave >> 1;     // 0..3 : 32-row slice
  const int wn   = wave & 1;      // 0..1 : 64-col slice
  const int l16  = lane & 15;
  const int half = lane >> 4;
  const int bm0  = blockIdx.y * BM;
  const int bn0  = blockIdx.x * BN;

  const v8f vzero = {};
  v8f acc[2][4];
#pragma unroll
  for (int i = 0; i < 2; ++i)
#pragma unroll
    for (int j = 0; j < 4; ++j) acc[i][j] = vzero;

  const int bk = tid >> 3, bn = (tid & 7) * 16;   // B: 32 k-rows x 8 col-groups
  const _Float16* Abase = A + (size_t)bm0 * K;
  const int nchunks = K / BK;

  // TDM: prime chunk 0 into buffer 0 (wave 0 only; EXEC-independent DMA)
  if (wave == 0)
    tdm_load_tile_f16(Abase, (unsigned)(uintptr_t)&As[0][0][0], BK, BM, (unsigned)K);

  for (int idx = 0; idx < nchunks; ++idx) {
    const int k0  = idx * BK;
    const int cur = idx & 1;

    // ---- stage B tile transposed into [n][k] (all threads)
    const _Float16* bp = Bm + (size_t)(k0 + bk) * N + bn0 + bn;
    v8h b0 = *(const v8h*)bp;
    v8h b1 = *(const v8h*)(bp + 8);
#pragma unroll
    for (int i = 0; i < 8; ++i) Bs[bn + i][bk] = b0[i];
#pragma unroll
    for (int i = 0; i < 8; ++i) Bs[bn + 8 + i][bk] = b1[i];
    if (k0 + BK < K)  // prefetch next B K-chunk (global_prefetch_b8)
      __builtin_prefetch((const void*)(bp + (size_t)BK * N), 0, 1);

    // ---- TDM: issue next A chunk into the other buffer, then wait for the
    //      current one (in-order TENSORcnt: <=1 outstanding => chunk idx done)
    if (wave == 0) {
      if (idx + 1 < nchunks) {
        tdm_load_tile_f16(Abase + (k0 + BK),
                          (unsigned)(uintptr_t)&As[cur ^ 1][0][0],
                          BK, BM, (unsigned)K);
        __builtin_amdgcn_s_wait_tensorcnt((short)1);  // chunk idx complete
      } else {
        __builtin_amdgcn_s_wait_tensorcnt((short)0);  // drain on last chunk
      }
    }
    __syncthreads();   // publish As[cur] (TDM) + Bs to all waves

    // ---- fragments from LDS
    v16h af[2], bf[4];
#pragma unroll
    for (int tm = 0; tm < 2; ++tm) {
      // A 16x32 layout: lane m=l16, K runs at half*8 and 16+half*8
      const _Float16* p = &As[cur][wm * 32 + tm * 16 + l16][half * 8];
      af[tm] = cat8(*(const v8h*)p, *(const v8h*)(p + 16));
    }
#pragma unroll
    for (int tn = 0; tn < 4; ++tn) {
      // B 32x16 layout: lane col n=l16, K runs at half*8 and 16+half*8
      const _Float16* p = &Bs[wn * 64 + tn * 16 + l16][half * 8];
      bf[tn] = cat8(*(const v8h*)p, *(const v8h*)(p + 16));
    }
#pragma unroll
    for (int tm = 0; tm < 2; ++tm)
#pragma unroll
      for (int tn = 0; tn < 4; ++tn)
        acc[tm][tn] = wmma16(af[tm], bf[tn], acc[tm][tn]);
    __syncthreads();   // all reads done before next B store / A TDM overwrite
  }

  // ---- epilogue: D layout -> global (row m = r + 8*half, col = l16)
#pragma unroll
  for (int tm = 0; tm < 2; ++tm)
#pragma unroll
    for (int tn = 0; tn < 4; ++tn) {
      const int gr0 = bm0 + wm * 32 + tm * 16;
      const int gc  = bn0 + wn * 64 + tn * 16 + l16;
#pragma unroll
      for (int r = 0; r < 8; ++r) {
        const int gr = gr0 + r + 8 * half;
        const float v = acc[tm][tn][r];
        if (out_f16)
          ((_Float16*)Out)[(size_t)gr * N + gc] = (_Float16)v;
        else
          ((float*)Out)[(size_t)gr * N + gc] = v;
      }
    }
}

// ---------------- sliding-window flash attention with WMMA ----------------
// grid = B*NH*(T/128); 8 waves/block; wave -> one 16-row query tile.
// 9 steps of 32 keys covering [q0-256, q0+31]; causal + window mask.
#define LPA 40        // f16 pitch for wave-private LDS staging
#define NSTEP (WIN_ / 32 + 1)   // 9

__global__ __launch_bounds__(256) void attn_kernel(
    const _Float16* __restrict__ qkv, _Float16* __restrict__ y) {
  __shared__ __align__(16) _Float16 plds[8][16][LPA];  // per-wave P (16x32)
  __shared__ __align__(16) _Float16 vlds[8][64][LPA];  // per-wave V^T (64 d x 32 key)

  const int tid  = threadIdx.x;
  const int lane = tid & 31;
  const int wave = tid >> 5;
  const int l16  = lane & 15;
  const int half = lane >> 4;

  const int blk  = blockIdx.x;
  const int qblk = blk & (T_ / 128 - 1);
  const int bh   = blk / (T_ / 128);
  const int h    = bh % NH_;
  const int b    = bh / NH_;
  const int q0   = qblk * 128 + wave * 16;

  const size_t rs = (size_t)C3_;                        // qkv row stride (f16)
  const _Float16* base = qkv + (size_t)b * T_ * rs;
  const int qoff = h * HS_;
  const int koff = C_ + h * HS_;
  const int voff = 2 * C_ + h * HS_;

  // q A-fragments for d-chunks 0..31 / 32..63, with 1/sqrt(HS)=0.125 folded in
  v16h aq[2];
#pragma unroll
  for (int c = 0; c < 2; ++c) {
    const _Float16* p = base + (size_t)(q0 + l16) * rs + qoff + c * 32 + half * 8;
    v16h t = cat8(*(const v8h*)p, *(const v8h*)(p + 16));
    aq[c] = t * (_Float16)0.125f;
  }

  const v8f vzero = {};
  v8f acc[4];  // 16 x 64 output accumulator (4 n-tiles)
#pragma unroll
  for (int nt = 0; nt < 4; ++nt) acc[nt] = vzero;
  float mrow[8], lrow[8];
#pragma unroll
  for (int r = 0; r < 8; ++r) { mrow[r] = -1e30f; lrow[r] = 0.f; }

  for (int s = 0; s < NSTEP; ++s) {
    const int jb = q0 - WIN_ + s * 32;   // base key index of this 32-key step

    // ---- S = q * k^T  (two 16-key tiles, each two d-chunks) ----
    v8f S[2];
#pragma unroll
    for (int x = 0; x < 2; ++x) {
      const int j  = jb + x * 16 + l16;          // this lane's key column
      const int jc = min(max(j, 0), T_ - 1);     // clamp address (masked later)
      v8f sacc = vzero;
#pragma unroll
      for (int c = 0; c < 2; ++c) {
        // B 32x16 frag: lane col = key j, K = d = c*32 + {half*8, 16+half*8} + e
        const _Float16* p = base + (size_t)jc * rs + koff + c * 32 + half * 8;
        v16h bk = cat8(*(const v8h*)p, *(const v8h*)(p + 16));
        sacc = wmma16(aq[c], bk, sacc);
      }
      S[x] = sacc;
    }

    // ---- mask + online softmax (row stats across 16-lane D-layout groups) ----
    float nmax[8];
#pragma unroll
    for (int r = 0; r < 8; ++r) {
      const int i  = q0 + r + 8 * half;
      const int j0 = jb + l16;
      const int j1 = jb + 16 + l16;
      float s0 = S[0][r], s1 = S[1][r];
      s0 = (j0 >= 0 && j0 <= i && (i - j0) < WIN_) ? s0 : -1e30f;
      s1 = (j1 >= 0 && j1 <= i && (i - j1) < WIN_) ? s1 : -1e30f;
      S[0][r] = s0; S[1][r] = s1;
      float v = fmaxf(s0, s1);
      v = fmaxf(v, __shfl_xor(v, 1, 16));
      v = fmaxf(v, __shfl_xor(v, 2, 16));
      v = fmaxf(v, __shfl_xor(v, 4, 16));
      v = fmaxf(v, __shfl_xor(v, 8, 16));
      nmax[r] = v;
    }
#pragma unroll
    for (int r = 0; r < 8; ++r) {
      const float mnew  = fmaxf(fmaxf(mrow[r], nmax[r]), -30.f);  // masked rows -> P=0
      const float alpha = __expf(mrow[r] - mnew);
      mrow[r] = mnew;
      const float p0 = __expf(S[0][r] - mnew);
      const float p1 = __expf(S[1][r] - mnew);
      S[0][r] = p0; S[1][r] = p1;
      float t = p0 + p1;
      t += __shfl_xor(t, 1, 16);
      t += __shfl_xor(t, 2, 16);
      t += __shfl_xor(t, 4, 16);
      t += __shfl_xor(t, 8, 16);
      lrow[r] = lrow[r] * alpha + t;
      acc[0][r] *= alpha; acc[1][r] *= alpha; acc[2][r] *= alpha; acc[3][r] *= alpha;
    }

    // ---- marshal P (D-layout) -> A-layout via wave-private LDS ----
#pragma unroll
    for (int r = 0; r < 8; ++r) {
      const int m = r + 8 * half;
      plds[wave][m][l16]      = (_Float16)S[0][r];
      plds[wave][m][16 + l16] = (_Float16)S[1][r];
    }
    // ---- stage V^T: lane loads v row (key jb+lane), scatters to [d][key] ----
    {
      const int jc = min(max(jb + lane, 0), T_ - 1);  // garbage rows see P==0
      const _Float16* p = base + (size_t)jc * rs + voff;
#pragma unroll
      for (int c = 0; c < 4; ++c) {
        v8h vv = *(const v8h*)(p + c * 8);
#pragma unroll
        for (int e = 0; e < 8; ++e) vlds[wave][c * 8 + e][lane] = vv[e];
      }
    }
    __syncthreads();  // uniform trip count across all 8 waves

    // ---- acc += P (16x32) * V (32x64) ----
    v16h pf;
    {
      const _Float16* p = &plds[wave][l16][half * 8];
      pf = cat8(*(const v8h*)p, *(const v8h*)(p + 16));
    }
#pragma unroll
    for (int nt = 0; nt < 4; ++nt) {
      const _Float16* p = &vlds[wave][nt * 16 + l16][half * 8];
      v16h vf = cat8(*(const v8h*)p, *(const v8h*)(p + 16));
      acc[nt] = wmma16(pf, vf, acc[nt]);
    }
    __syncthreads();
  }

  // ---- normalize + write y (f16) ----
  float inv[8];
#pragma unroll
  for (int r = 0; r < 8; ++r) inv[r] = 1.0f / lrow[r];
#pragma unroll
  for (int nt = 0; nt < 4; ++nt)
#pragma unroll
    for (int r = 0; r < 8; ++r) {
      const int i   = q0 + r + 8 * half;
      const int col = h * HS_ + nt * 16 + l16;
      y[(size_t)(b * T_ + i) * C_ + col] = (_Float16)(acc[nt][r] * inv[r]);
    }
}

// ---------------- host-side orchestration ----------------
extern "C" void kernel_launch(void* const* d_in, const int* in_sizes, int n_in,
                              void* d_out, int out_size, void* d_ws, size_t ws_size,
                              hipStream_t stream) {
  const float* x  = (const float*)d_in[0];
  const float* Wa = (const float*)d_in[1];
  const float* Wp = (const float*)d_in[2];
  float* out = (float*)d_out;

  // workspace (f16): x | W_attn | W_proj | qkv | y   == 48 MiB total
  _Float16* xh   = (_Float16*)d_ws;
  _Float16* Wah  = xh   + (size_t)M_ * C_;
  _Float16* Wph  = Wah  + (size_t)C_ * C3_;
  _Float16* qkvh = Wph  + (size_t)C_ * C_;
  _Float16* yh   = qkvh + (size_t)M_ * C3_;

  cvt_kernel<<<1024, 256, 0, stream>>>(x,  xh,  M_ * C_);
  cvt_kernel<<<1024, 256, 0, stream>>>(Wa, Wah, C_ * C3_);
  cvt_kernel<<<1024, 256, 0, stream>>>(Wp, Wph, C_ * C_);

  // qkv = x @ W_attn   (4096 x 3072 x 1024), f16 out
  gemm_kernel<<<dim3(C3_ / BN, M_ / BM), 256, 0, stream>>>(
      xh, Wah, (void*)qkvh, M_, C3_, C_, 1);

  // sliding-window attention -> y (f16)
  attn_kernel<<<B_ * NH_ * (T_ / 128), 256, 0, stream>>>(qkvh, yh);

  // out = y @ W_proj   (4096 x 1024 x 1024), f32 out
  gemm_kernel<<<dim3(C_ / BN, M_ / BM), 256, 0, stream>>>(
      yh, Wph, (void*)out, M_, C_, C_, 0);

  (void)in_sizes; (void)n_in; (void)out_size; (void)ws_size;
}
